// GraphConvPooling_29892972380764
// MI455X (gfx1250) — matrix operand order repaired and checked by reading, not compiled, not run on hardware
//
#include <hip/hip_runtime.h>
#include <stdint.h>

// Problem constants (from the reference)
#define BGRAPH 64
#define NHITS  1024
#define DIM    64
#define ROWS_PER_BLOCK 64          // 4 waves x 16 rows
#define WAVES   4
#define THREADS 128
#define KCHUNK  32                 // K step = one WMMA K (bf16) = one mask word
#define NKC     (NHITS / KCHUNK)   // 32 K-chunks over the 1024 neighbor dim
#define MASK_WORDS ((size_t)BGRAPH * NHITS * NKC)   // 2,097,152 u32 = 8 MB
#define XTG_ELEMS  ((size_t)BGRAPH * DIM * NHITS)   // bf16 transposed x, 8 MB

typedef __attribute__((ext_vector_type(16))) __bf16 v16bf;
typedef __attribute__((ext_vector_type(8)))  float  v8f;
typedef __attribute__((ext_vector_type(4)))  unsigned int v4u;
typedef __attribute__((ext_vector_type(8)))  int    v8i;
typedef __attribute__((ext_vector_type(4)))  int    v4i;

#if __has_builtin(__builtin_amdgcn_tensor_load_to_lds) && __has_builtin(__builtin_amdgcn_s_wait_tensorcnt)
#define HAS_TDM 1
#else
#define HAS_TDM 0
#endif

union Frag { v16bf bf; unsigned int u[8]; };

__device__ __forceinline__ unsigned short f2bf(float f) {
  unsigned int u = __float_as_uint(f);
  unsigned int r = u + 0x7FFFu + ((u >> 16) & 1u);   // round-to-nearest-even
  return (unsigned short)(r >> 16);
}

__device__ __forceinline__ void fill8(Frag& f, uint4 lo, uint4 hi) {
  f.u[0] = lo.x; f.u[1] = lo.y; f.u[2] = lo.z; f.u[3] = lo.w;
  f.u[4] = hi.x; f.u[5] = hi.y; f.u[6] = hi.z; f.u[7] = hi.w;
}

#if HAS_TDM
// TDM: DMA a 64(n) x 32(k) bf16 tile of XTg[b] (row stride 1024 elems) into LDS,
// packed row-major [n][32] -- exactly the XT layout the B-fragment loads expect.
__device__ __forceinline__ void tdm_load_chunk(const unsigned short* gsrc, unsigned lds_off) {
  unsigned long long ga = (unsigned long long)(size_t)gsrc;
  // D# group0: count=1 | lds_addr | global_addr[56:0] | type=2 ("image")
  v4u g0 = { 1u, lds_off, (unsigned)ga,
             ((unsigned)((ga >> 32) & 0x01FFFFFFull)) | 0x80000000u };
  // D# group1: data_size=1 (2B) @ [17:16]; tensor_dim0=1024 @ [79:48];
  // tensor_dim1=64 @ [111:80]; tile_dim0=32 @ [127:112]; tile_dim1=64 @ [143:128];
  // tensor_dim0_stride=1024 @ [207:160]
  v8i g1 = { (int)0x00010000u,          // wg_mask=0, data_size=1(2B)
             (int)(1024u << 16),        // tensor_dim0 lo16 in [63:48]
             (int)(64u << 16),          // tensor_dim0 hi16=0, tensor_dim1 lo16
             (int)(32u << 16),          // tensor_dim1 hi16=0, tile_dim0=32
             (int)64,                   // tile_dim1=64, tile_dim2=0
             (int)1024,                 // tensor_dim0_stride lo32
             0, 0 };                    // stride hi, tensor_dim1_stride (unused, 2D)
  v4i z4 = { 0, 0, 0, 0 };
  v8i z8 = { 0, 0, 0, 0, 0, 0, 0, 0 };
  __builtin_amdgcn_tensor_load_to_lds(g0, g1, z4, z4, z8, 0);  // 6-arg toolchain
}
#endif

// ---------------- kernel 0: zero the adjacency bitmask -----------------
__global__ __launch_bounds__(256) void zero_mask_kernel(unsigned int* __restrict__ m) {
  size_t i = (size_t)blockIdx.x * blockDim.x + threadIdx.x;  // 524288 threads x 16B
  uint4 z = {0u, 0u, 0u, 0u};
  reinterpret_cast<uint4*>(m)[i] = z;
}

// ---------------- kernel 1: scatter edges into bitmask (OR = dedup) ----
__global__ __launch_bounds__(256) void edge_kernel(const int* __restrict__ src,
                                                   const int* __restrict__ dst,
                                                   unsigned int* __restrict__ mask,
                                                   int E) {
  int e = blockIdx.x * blockDim.x + threadIdx.x;
  if (e >= E) return;
  int row = src[e];                 // global node id = b*NHITS + s_loc
  int col = dst[e] & (NHITS - 1);   // local column within graph
  atomicOr(&mask[((size_t)row << 5) + (col >> 5)], 1u << (col & 31));
}

// ------- kernel 1b: convert x f32 -> bf16, transposed per graph --------
// XTg[b][n][k] with k (node-within-graph) contiguous; packed 4xbf16 stores.
__global__ __launch_bounds__(256) void xconv_kernel(const float* __restrict__ x,
                                                    unsigned short* __restrict__ xtg) {
  int tid = blockIdx.x * 256 + threadIdx.x;        // 1,048,576 threads
  int b   = tid >> 14;                             // 16384 quads per graph
  int rem = tid & 16383;
  int n   = rem & 63;                              // lane-adjacent n -> coalesced reads
  int k4  = rem >> 6;                              // 0..255
  int k   = k4 << 2;
  const float* xp = x + ((size_t)(b * NHITS + k)) * DIM + n;
  unsigned long long pack = 0;
  #pragma unroll
  for (int j = 0; j < 4; ++j)
    pack |= (unsigned long long)f2bf(xp[(size_t)j * DIM]) << (16 * j);
  reinterpret_cast<unsigned long long*>(xtg)[(((size_t)(b * DIM + n)) << 8) + k4] = pack;
}

// ---------------- kernel 2: fused (A@x)@W + bias, row-tile max ---------
__global__ __launch_bounds__(THREADS) void gcn_main_kernel(
    const unsigned short* __restrict__ xtg, // [B][DIM][NHITS] bf16 transposed x
    const unsigned int* __restrict__ mask,  // [B*NHITS][NKC] bitmask rows
    const float* __restrict__ W,            // [DIM, DIM] f32
    const float* __restrict__ bias,         // [DIM]
    float* __restrict__ pool_partial)       // [B*16, DIM]
{
  __shared__ unsigned int maskS[ROWS_PER_BLOCK][NKC + 1];              // 8448 B
  __shared__ __align__(16) unsigned short XT[2][DIM][KCHUNK];          // 8 KB double buffer
  __shared__ __align__(16) unsigned short WT[DIM][72];                 // 9216 B, WT[n][k]=W[k][n]
  __shared__ __align__(16) unsigned short AGG[WAVES][16][72];          // 9216 B
  __shared__ float wmaxS[WAVES][DIM];                                  // 1 KB

  const int b       = blockIdx.x >> 4;
  const int tile    = blockIdx.x & 15;
  const int rowbase = tile * ROWS_PER_BLOCK;
  const int t     = threadIdx.x;
  const int wave  = t >> 5;
  const int lane  = t & 31;
  const int lhalf = lane & 15;
  const bool hi   = (lane >= 16);

  const unsigned short* xtg_b = xtg + (size_t)b * DIM * NHITS;

#if HAS_TDM
  if (wave == 0) tdm_load_chunk(xtg_b, (unsigned)(size_t)&XT[0][0][0]);
#endif

  // Preload this block's 64 mask rows (64*32 words, coalesced).
  #pragma unroll
  for (int i = 0; i < (ROWS_PER_BLOCK * NKC) / THREADS; ++i) {
    int e = t + i * THREADS;
    int r = e >> 5, kc = e & 31;
    maskS[r][kc] = mask[(((size_t)b * NHITS + rowbase + r) << 5) + kc];
  }
  // Preload W transposed as bf16 into LDS.
  #pragma unroll
  for (int i = 0; i < (DIM * DIM) / THREADS; ++i) {
    int e = t + i * THREADS;
    int k = e >> 6, n = e & 63;
    WT[n][k] = f2bf(W[k * DIM + n]);
  }

  v8f acc[4] = {};   // 16 rows x 64 cols of agg = A_tile @ X

  for (int kc32 = 0; kc32 < NKC; ++kc32) {
    const int cur = kc32 & 1;
    __syncthreads();  // everyone done with the buffer about to be refilled
#if HAS_TDM
    if (wave == 0) {
      if (kc32 + 1 < NKC) {  // DMA next chunk into the other buffer while we compute
        tdm_load_chunk(xtg_b + (size_t)(kc32 + 1) * KCHUNK,
                       (unsigned)(size_t)&XT[cur ^ 1][0][0]);
        __builtin_amdgcn_s_wait_tensorcnt(1);  // chunk kc32 landed; kc32+1 in flight
      } else {
        __builtin_amdgcn_s_wait_tensorcnt(0);  // last chunk landed
      }
    }
#else
    {  // fallback: cooperative copy of chunk kc32 (bf16, 512 dwords)
      const unsigned int* srcu = reinterpret_cast<const unsigned int*>(xtg_b);
      unsigned int* dstu = reinterpret_cast<unsigned int*>(&XT[cur][0][0]);
      #pragma unroll
      for (int i = 0; i < 512 / THREADS; ++i) {
        int e = t + i * THREADS;
        int n = e >> 4, r32 = e & 15;
        dstu[e] = srcu[(size_t)n * (NHITS / 2) + kc32 * 16 + r32];
      }
    }
#endif
    __syncthreads();

    // A fragment: expand this wave's 16 rows' mask word into bf16 {0,1}
    // per the 16-bit A 16x32 layout (lanes 0-15: K0-7,K16-23; lanes 16-31: K8-15,K24-31).
    unsigned int w = maskS[(wave << 4) + lhalf][kc32];
    const int base = hi ? 8 : 0;
    Frag a;
    #pragma unroll
    for (int i = 0; i < 4; ++i) {
      unsigned int b0 = (w >> (base + 2 * i)) & 1u;
      unsigned int b1 = (w >> (base + 2 * i + 1)) & 1u;
      a.u[i] = (b0 ? 0x00003F80u : 0u) | (b1 ? 0x3F800000u : 0u);
      unsigned int c0 = (w >> (base + 16 + 2 * i)) & 1u;
      unsigned int c1 = (w >> (base + 16 + 2 * i + 1)) & 1u;
      a.u[4 + i] = (c0 ? 0x00003F80u : 0u) | (c1 ? 0x3F800000u : 0u);
    }

    // 4 column tiles of X: B fragment = 2x ds_load_b128 from the DMA'd tile.
    #pragma unroll
    for (int nt = 0; nt < 4; ++nt) {
      const unsigned short* p = &XT[cur][nt * 16 + lhalf][hi ? 8 : 0];
      Frag bf;
      fill8(bf, *reinterpret_cast<const uint4*>(p),
                *reinterpret_cast<const uint4*>(p + 16));
      acc[nt] = __builtin_amdgcn_wmma_f32_16x16x32_bf16(
          false, a.bf, false, bf.bf, (short)0, acc[nt], false, false);
    }
  }

  // Restage agg (16x64 f32 accumulators) as bf16 in LDS, A-matrix layout source.
  #pragma unroll
  for (int nt = 0; nt < 4; ++nt)
    #pragma unroll
    for (int v = 0; v < 8; ++v) {
      int m = v + (hi ? 8 : 0);
      AGG[wave][m][nt * 16 + lhalf] = f2bf(acc[nt][v]);
    }
  // per-wave private LDS region; DS ops from the same wave stay in order.

  // Second GEMM: Y(16x64) = AGG(16x64) @ W(64x64)
  v8f yacc[4] = {};
  #pragma unroll
  for (int kc = 0; kc < 2; ++kc) {
    const unsigned short* pa = &AGG[wave][lhalf][kc * 32 + (hi ? 8 : 0)];
    Frag a2;
    fill8(a2, *reinterpret_cast<const uint4*>(pa),
              *reinterpret_cast<const uint4*>(pa + 16));
    #pragma unroll
    for (int nt = 0; nt < 4; ++nt) {
      const unsigned short* pb = &WT[nt * 16 + lhalf][kc * 32 + (hi ? 8 : 0)];
      Frag b2;
      fill8(b2, *reinterpret_cast<const uint4*>(pb),
                *reinterpret_cast<const uint4*>(pb + 16));
      yacc[nt] = __builtin_amdgcn_wmma_f32_16x16x32_bf16(
          false, a2.bf, false, b2.bf, (short)0, yacc[nt], false, false);
    }
  }

  // bias + max over this wave's 16 rows (VGPRs cover 8 rows, lane-halves the other 8)
  #pragma unroll
  for (int nt = 0; nt < 4; ++nt) {
    float bn = bias[nt * 16 + lhalf];
    float m = -3.0e38f;
    #pragma unroll
    for (int v = 0; v < 8; ++v) m = fmaxf(m, yacc[nt][v] + bn);
    m = fmaxf(m, __shfl_xor(m, 16, 32));
    if (!hi) wmaxS[wave][nt * 16 + lhalf] = m;
  }
  __syncthreads();
  if (t < DIM) {
    float m = wmaxS[0][t];
    #pragma unroll
    for (int wv = 1; wv < WAVES; ++wv) m = fmaxf(m, wmaxS[wv][t]);
    pool_partial[(size_t)blockIdx.x * DIM + t] = m;
  }
}

// ---------------- kernel 3: combine partial maxima + MLP ----------------
__global__ __launch_bounds__(64) void gcn_final_kernel(
    const float* __restrict__ pp,
    const float* __restrict__ w1, const float* __restrict__ b1,
    const float* __restrict__ w2, const float* __restrict__ b2,
    const float* __restrict__ w3, const float* __restrict__ b3,
    float* __restrict__ out)
{
  __shared__ float pooled[DIM], h1[DIM], h2[DIM], red[DIM];
  const int b = blockIdx.x, n = threadIdx.x;

  float m = pp[(size_t)(b * 16) * DIM + n];
  #pragma unroll
  for (int tl = 1; tl < 16; ++tl)
    m = fmaxf(m, pp[(size_t)(b * 16 + tl) * DIM + n]);
  pooled[n] = m;
  __syncthreads();

  float s = b1[n];
  for (int k = 0; k < DIM; ++k) s += pooled[k] * w1[k * DIM + n];
  h1[n] = tanhf(s);
  __syncthreads();

  s = b2[n];
  for (int k = 0; k < DIM; ++k) s += h1[k] * w2[k * DIM + n];
  h2[n] = tanhf(s);
  __syncthreads();

  red[n] = h2[n] * w3[n];
  __syncthreads();
  for (int off = 32; off >= 1; off >>= 1) {
    if (n < off) red[n] += red[n + off];
    __syncthreads();
  }
  if (n == 0) out[b] = red[0] + b3[0];
}

// ------------------------------ launcher --------------------------------
extern "C" void kernel_launch(void* const* d_in, const int* in_sizes, int n_in,
                              void* d_out, int out_size, void* d_ws, size_t ws_size,
                              hipStream_t stream) {
  (void)n_in; (void)out_size; (void)ws_size;
  const float* x    = (const float*)d_in[0];
  const int*   eidx = (const int*)d_in[1];     // [2, E] int32
  // d_in[2] = batch (derivable), d_in[3] = batch_size (fixed 64): unused
  const float* W    = (const float*)d_in[4];
  const float* bias = (const float*)d_in[5];
  const float* w1   = (const float*)d_in[6];
  const float* b1   = (const float*)d_in[7];
  const float* w2   = (const float*)d_in[8];
  const float* b2   = (const float*)d_in[9];
  const float* w3   = (const float*)d_in[10];
  const float* b3   = (const float*)d_in[11];
  float* out = (float*)d_out;

  unsigned int*   mask = (unsigned int*)d_ws;                        // 8 MB bitmask
  unsigned short* xtg  = (unsigned short*)((char*)d_ws + MASK_WORDS * 4);  // 8 MB bf16 x^T
  float* pool = (float*)((char*)d_ws + MASK_WORDS * 4 + XTG_ELEMS * 2);    // 256 KB

  const int E = in_sizes[1] / 2;
  const int* src = eidx;
  const int* dst = eidx + E;

  zero_mask_kernel<<<(int)(MASK_WORDS / 4 / 256), 256, 0, stream>>>(mask);
  edge_kernel<<<(E + 255) / 256, 256, 0, stream>>>(src, dst, mask, E);
  xconv_kernel<<<(int)(XTG_ELEMS / 4 / 256), 256, 0, stream>>>(x, xtg);
  gcn_main_kernel<<<BGRAPH * 16, THREADS, 0, stream>>>(xtg, mask, W, bias, pool);
  gcn_final_kernel<<<BGRAPH, 64, 0, stream>>>(pool, w1, b1, w2, b2, w3, b3, out);
}
